// PairEncoder_43087111913645
// MI455X (gfx1250) — matrix operand order repaired and checked by reading, not compiled, not run on hardware
//
#include <hip/hip_runtime.h>
#include <cstdint>
#include <cstddef>

#define PLEN 400
#define QLEN 60
#define BATCH 128
#define PIN 512
#define QIN 512
#define HID 256
#define DD 1024
#define G3 768

typedef __attribute__((ext_vector_type(16))) __bf16 v16bf;
typedef __attribute__((ext_vector_type(8)))  float  v8f;

// ---------------------------------------------------------------------------
// Fast transcendentals on hardware TRANS pipes (v_exp_f32 / v_rcp_f32).
// tanh(x) = 1 - 2/(e^{2x}+1); sig(x) = 1/(1+e^{-x}). Inf/0 cases saturate
// correctly through v_rcp (rcp(inf)=0).
// ---------------------------------------------------------------------------
__device__ __forceinline__ float fast_tanh(float x) {
  return 1.f - 2.f * __builtin_amdgcn_rcpf(__expf(2.f * x) + 1.f);
}
__device__ __forceinline__ float fast_sig(float x) {
  return __builtin_amdgcn_rcpf(1.f + __expf(-x));
}

// ---------------------------------------------------------------------------
// WMMA helpers (CDNA5 wave32, 16x16x32 bf16 -> f32)
// ---------------------------------------------------------------------------
__device__ __forceinline__ v8f wmma_bf16(v16bf a, v16bf b, v8f c) {
  return __builtin_amdgcn_wmma_f32_16x16x32_bf16(false, a, false, b, (short)0, c,
                                                 false, false);
}

// A-matrix 16x32 bf16 layout (ISA 7.12.2): lanes 0-15 hold K 0..7,16..23 of
// row M=lane; lanes 16-31 hold K 8..15,24..31 of row M=lane-16.
__device__ __forceinline__ v16bf load_A_bf(const __bf16* A, int lda, int m0,
                                           int k0, int lane) {
  int m = m0 + (lane & 15);
  int kg = (lane >> 4) << 3;  // 0 or 8
  const __bf16* r = A + (size_t)m * lda + k0 + kg;
  v16bf a;
#pragma unroll
  for (int i = 0; i < 8; ++i) { a[i] = r[i]; a[i + 8] = r[i + 16]; }
  return a;
}

// Same, with on-the-fly fp32 -> bf16 conversion (fresh activations each step).
__device__ __forceinline__ v16bf load_A_f32(const float* A, int lda, int m0,
                                            int k0, int lane) {
  int m = m0 + (lane & 15);
  int kg = (lane >> 4) << 3;
  const float* r = A + (size_t)m * lda + k0 + kg;
  v16bf a;
#pragma unroll
  for (int i = 0; i < 8; ++i) {
    a[i]     = (__bf16)r[i];
    a[i + 8] = (__bf16)r[i + 16];
  }
  return a;
}

// B operands are pre-packed lane-major: tile (kt,nt) stores, for each lane,
// its 16 bf16 values contiguously -> one 32B vector load per lane per WMMA.
__device__ __forceinline__ v16bf load_B_packed(const __bf16* Bp, int ntiles,
                                               int kt, int nt, int lane) {
  const __bf16* p = Bp + ((((size_t)kt * ntiles + nt) << 5) + lane) * 16;
  return *(const v16bf*)p;
}

// C/D 16x16 f32 layout: VGPR i -> row (i or 8+i), lane -> column.
__device__ __forceinline__ void store_C(float* C, int ldc, int m0, int n0,
                                        int lane, v8f acc) {
  int n = n0 + (lane & 15);
  int mr = m0 + ((lane >> 4) << 3);
#pragma unroll
  for (int i = 0; i < 8; ++i) C[(size_t)(mr + i) * ldc + n] = acc[i];
}

// ---------------------------------------------------------------------------
// Global (grid-wide) barrier for the persistent scan kernel. Monotonic
// counter + generation; counters reset each launch via hipMemsetAsync.
// ---------------------------------------------------------------------------
__device__ __forceinline__ void grid_sync(unsigned* bar, unsigned nwg) {
  __syncthreads();
  if (threadIdx.x == 0) {
    __threadfence();
    unsigned old_gen = __atomic_load_n(&bar[1], __ATOMIC_RELAXED);
    unsigned arrive = atomicAdd(&bar[0], 1u) + 1u;
    if (arrive % nwg == 0u) {
      atomicAdd(&bar[1], 1u);
    } else {
      while (__atomic_load_n(&bar[1], __ATOMIC_RELAXED) == old_gen)
        __builtin_amdgcn_s_sleep(2);
    }
    __threadfence();
  }
  __syncthreads();
}

// ---------------------------------------------------------------------------
// Weight repack: fp32 [K][N] or [N][K] -> bf16 lane-major WMMA tiles.
// ---------------------------------------------------------------------------
__global__ void pack_B_kernel(const float* __restrict__ src,
                              __bf16* __restrict__ dst, int K, int N,
                              int transpose) {
  int idx = blockIdx.x * blockDim.x + threadIdx.x;
  int total = K * N;
  if (idx >= total) return;
  int tile = idx >> 9;          // 512 elements per 32x16 tile
  int rem = idx & 511;
  int lane = rem >> 4, i = rem & 15;
  int ntiles = N >> 4;
  int kt = tile / ntiles, nt = tile - kt * ntiles;
  int k = kt * 32 + ((lane >> 4) << 3) + (i & 7) + ((i >> 3) << 4);  // lanes 0-15: K 0..7,16..23; 16-31: 8..15,24..31
  int n = nt * 16 + (lane & 15);
  float v = transpose ? src[(size_t)n * K + k] : src[(size_t)k * N + n];
  dst[idx] = (__bf16)v;
}

// ---------------------------------------------------------------------------
// q_temp = q_input @ Wq (per direction): (60*128, 512) x (512, 256)
// ---------------------------------------------------------------------------
__global__ __launch_bounds__(256) void qtemp_kernel(
    const float* __restrict__ q_input, const __bf16* __restrict__ wqB,
    float* __restrict__ qtemp) {
  int gw = (int)((blockIdx.x * blockDim.x + threadIdx.x) >> 5);
  int lane = threadIdx.x & 31;
  const int MT = (QLEN * BATCH) / 16;  // 480
  const int NT = HID / 16;             // 16
  const int TILES = MT * NT;           // 7680
  if (gw >= 2 * TILES) return;
  int dir = gw / TILES;
  int t = gw - dir * TILES;
  int mt = t / NT, nt = t - mt * NT;
  const __bf16* bq = wqB + (size_t)dir * QIN * HID;
  v8f acc = {};
  for (int kt = 0; kt < QIN / 32; ++kt) {
    v16bf a = load_A_f32(q_input, QIN, mt * 16, kt * 32, lane);
    v16bf b = load_B_packed(bq, NT, kt, nt, lane);
    acc = wmma_bf16(a, b, acc);
  }
  store_C(qtemp + (size_t)dir * QLEN * BATCH * HID, HID, mt * 16, nt * 16, lane,
          acc);
}

// ---------------------------------------------------------------------------
// Persistent scan kernel: 32 WGs (16 per direction, 8 wave32 each).
// Per step: A) state@Ws + p@Wp  B) attention + gate-input  C) gate GEMM + gh
//           E) gi GEMM  F) GRU elementwise; phases split by grid barriers.
// All weights live in L2 in packed bf16; Ws tiles are staged in 128KB LDS
// (CDNA5: 320KB/WGP) so the per-step recurrent GEMM reads via ds_load.
// ---------------------------------------------------------------------------
__global__ __launch_bounds__(256) void scan_kernel(
    const float* __restrict__ q_input, const float* __restrict__ p_input,
    const unsigned char* __restrict__ q_mask, const float* __restrict__ v_all,
    const float* __restrict__ gate_b_all, const float* __restrict__ bih_all,
    const float* __restrict__ bhh_all, const __bf16* __restrict__ gateWT,
    const __bf16* __restrict__ wihT, const __bf16* __restrict__ whhT,
    const __bf16* __restrict__ wsB, const __bf16* __restrict__ wpB,
    const float* __restrict__ qtemp, float* __restrict__ stemp,
    float* __restrict__ ptmp, float* __restrict__ xpre,
    __bf16* __restrict__ xprebf, __bf16* __restrict__ xbf,
    float* __restrict__ gh, float* __restrict__ gi, float* __restrict__ state,
    __bf16* __restrict__ statebf, float* __restrict__ out,
    unsigned* __restrict__ bar) {
  const int NWG = 32, WPD = 16;  // total WGs, WGs per direction
  int wg = blockIdx.x;
  int dir = wg / WPD;
  int lwg = wg - dir * WPD;
  int tid = threadIdx.x;
  int lane = tid & 31;
  int warp = tid >> 5;
  int wid = lwg * 8 + warp;  // wave id within direction: 0..127

  const float* qt_d = qtemp + (size_t)dir * QLEN * BATCH * HID;
  const float* v_d = v_all + dir * HID;
  const float* gb_d = gate_b_all + dir * DD;
  const float* bih_d = bih_all + dir * G3;
  const float* bhh_d = bhh_all + dir * G3;
  const __bf16* gW_d = gateWT + (size_t)dir * DD * DD;
  const __bf16* wih_d = wihT + (size_t)dir * DD * G3;
  const __bf16* whh_d = whhT + (size_t)dir * HID * G3;
  const __bf16* ws_d = wsB + (size_t)dir * HID * HID;
  const __bf16* wp_d = wpB + (size_t)dir * PIN * HID;
  float* stemp_d = stemp + dir * BATCH * HID;
  float* ptmp_d = ptmp + dir * BATCH * HID;
  float* xpre_d = xpre + dir * BATCH * DD;
  __bf16* xprebf_d = xprebf + dir * BATCH * DD;
  __bf16* xbf_d = xbf + dir * BATCH * DD;
  float* gh_d = gh + dir * BATCH * G3;
  float* gi_d = gi + dir * BATCH * G3;
  float* st_d = state + dir * BATCH * HID;
  __bf16* stb_d = statebf + dir * BATCH * HID;

  // Stage packed Ws tiles for this direction into LDS (4096 x 32B fragments).
  __shared__ v16bf lds_ws[(HID / 32) * (HID / 16) * 32];  // 128 KB
  {
    const v16bf* g = (const v16bf*)ws_d;
    for (int i = tid; i < (HID / 32) * (HID / 16) * 32; i += 256)
      lds_ws[i] = g[i];
  }

  // init = zeros (workspace is poisoned between calls)
  for (int i = lwg * 256 + tid; i < BATCH * HID; i += WPD * 256) {
    st_d[i] = 0.f;
    stb_d[i] = (__bf16)0.f;
  }
  __syncthreads();
  grid_sync(bar, NWG);

  for (int step = 0; step < PLEN; ++step) {
    int p = dir ? (PLEN - 1 - step) : step;
    const float* px = p_input + (size_t)p * BATCH * PIN;

    // ---- Phase A: stemp = state@Ws (B from LDS); ptmp = p_x@Wp ----------
    {
      int mt = wid >> 4, nt = wid & 15;  // one 16x16 tile each, per GEMM
      v8f acc = {};
      for (int kt = 0; kt < HID / 32; ++kt) {
        v16bf a = load_A_bf(stb_d, HID, mt * 16, kt * 32, lane);
        v16bf b = lds_ws[((kt * (HID / 16) + nt) << 5) + lane];
        acc = wmma_bf16(a, b, acc);
      }
      store_C(stemp_d, HID, mt * 16, nt * 16, lane, acc);

      v8f acc2 = {};
      for (int kt = 0; kt < PIN / 32; ++kt) {
        v16bf a = load_A_f32(px, PIN, mt * 16, kt * 32, lane);
        v16bf b = load_B_packed(wp_d, HID / 16, kt, nt, lane);
        acc2 = wmma_bf16(a, b, acc2);
      }
      store_C(ptmp_d, HID, mt * 16, nt * 16, lane, acc2);
    }
    grid_sync(bar, NWG);

    // ---- Phase B: attention + build gated-GRU input (1 batch col / wave) --
    {
      int b = wid;
      const float* prow = ptmp_d + b * HID;
      const float* srow = stemp_d + b * HID;
      float l0 = -3.0e38f, l1 = -3.0e38f;
      int q0 = lane;
      if (q0 < QLEN) {
        const float* qrow = qt_d + ((size_t)q0 * BATCH + b) * HID;
        float s = 0.f;
        for (int h = 0; h < HID; ++h)
          s += fast_tanh(qrow[h] + prow[h] + srow[h]) * v_d[h];
        l0 = q_mask[q0 * BATCH + b] ? -3.0e38f : s;
      }
      int q1 = lane + 32;
      if (q1 < QLEN) {
        const float* qrow = qt_d + ((size_t)q1 * BATCH + b) * HID;
        float s = 0.f;
        for (int h = 0; h < HID; ++h)
          s += fast_tanh(qrow[h] + prow[h] + srow[h]) * v_d[h];
        l1 = q_mask[q1 * BATCH + b] ? -3.0e38f : s;
      }
      float m = fmaxf(l0, l1);
      for (int o = 16; o > 0; o >>= 1) m = fmaxf(m, __shfl_down(m, o, 32));
      m = __shfl(m, 0, 32);
      float e0 = (q0 < QLEN) ? __expf(l0 - m) : 0.f;
      float e1 = (q1 < QLEN) ? __expf(l1 - m) : 0.f;
      float ssum = e0 + e1;
      for (int o = 16; o > 0; o >>= 1) ssum += __shfl_down(ssum, o, 32);
      float rs = __builtin_amdgcn_rcpf(__shfl(ssum, 0, 32));
      float w0 = e0 * rs, w1 = e1 * rs;

      float4 attv[4];
#pragma unroll
      for (int j = 0; j < 4; ++j) attv[j] = make_float4(0.f, 0.f, 0.f, 0.f);
      for (int q = 0; q < QLEN; ++q) {
        float wq = (q < 32) ? __shfl(w0, q, 32) : __shfl(w1, q - 32, 32);
        const float4* qin4 =
            (const float4*)(q_input + ((size_t)q * BATCH + b) * QIN);
#pragma unroll
        for (int j = 0; j < 4; ++j) {
          float4 t = qin4[j * 32 + lane];
          attv[j].x += wq * t.x;
          attv[j].y += wq * t.y;
          attv[j].z += wq * t.z;
          attv[j].w += wq * t.w;
        }
      }
      const float4* px4 = (const float4*)(px + (size_t)b * PIN);
      float4* xp4 = (float4*)(xpre_d + (size_t)b * DD);
      __bf16* xb = xprebf_d + (size_t)b * DD;
#pragma unroll
      for (int j = 0; j < 4; ++j) {
        int e4 = j * 32 + lane;
        float4 pv = px4[e4];
        xp4[e4] = pv;
        xp4[(PIN / 4) + e4] = attv[j];
        int e = e4 * 4;
        xb[e] = (__bf16)pv.x; xb[e + 1] = (__bf16)pv.y;
        xb[e + 2] = (__bf16)pv.z; xb[e + 3] = (__bf16)pv.w;
        xb[PIN + e] = (__bf16)attv[j].x; xb[PIN + e + 1] = (__bf16)attv[j].y;
        xb[PIN + e + 2] = (__bf16)attv[j].z; xb[PIN + e + 3] = (__bf16)attv[j].w;
      }
    }
    grid_sync(bar, NWG);

    // ---- Phase C: gate GEMM (4 N-tiles/wave, shared A fragment) + gh ----
    {
      int mt = wid >> 4;
      int nt0 = (wid & 15) << 2;  // 4 consecutive N tiles of 64
      v8f acc[4] = {{}, {}, {}, {}};
      for (int kt = 0; kt < DD / 32; ++kt) {
        v16bf a = load_A_bf(xprebf_d, DD, mt * 16, kt * 32, lane);
        if (kt + 1 < DD / 32)
          __builtin_prefetch(gW_d + (((size_t)(kt + 1) * (DD / 16) + nt0) << 9),
                             0, 1);
#pragma unroll
        for (int j = 0; j < 4; ++j) {
          v16bf bfr = load_B_packed(gW_d, DD / 16, kt, nt0 + j, lane);
          acc[j] = wmma_bf16(a, bfr, acc[j]);
        }
      }
      int mr = mt * 16 + ((lane >> 4) << 3);
#pragma unroll
      for (int j = 0; j < 4; ++j) {
        int n = (nt0 + j) * 16 + (lane & 15);
        float gbn = gb_d[n];
#pragma unroll
        for (int i = 0; i < 8; ++i) {
          float xp = xpre_d[(size_t)(mr + i) * DD + n];
          float g = fast_sig(acc[j][i] + gbn);
          xbf_d[(size_t)(mr + i) * DD + n] = (__bf16)(xp * g);
        }
      }

      // gh = state @ Whh^T : 8x48 tiles, 3 per wave
      int nh0 = (wid & 15) * 3;
      v8f acch[3] = {{}, {}, {}};
      for (int kt = 0; kt < HID / 32; ++kt) {
        v16bf a = load_A_bf(stb_d, HID, mt * 16, kt * 32, lane);
#pragma unroll
        for (int j = 0; j < 3; ++j) {
          v16bf bfr = load_B_packed(whh_d, G3 / 16, kt, nh0 + j, lane);
          acch[j] = wmma_bf16(a, bfr, acch[j]);
        }
      }
#pragma unroll
      for (int j = 0; j < 3; ++j)
        store_C(gh_d, G3, mt * 16, (nh0 + j) * 16, lane, acch[j]);
    }
    grid_sync(bar, NWG);

    // ---- Phase E: gi = x_gated @ Wih^T : 8x48 tiles, 3 per wave ---------
    {
      int mt = wid >> 4;
      int nt0 = (wid & 15) * 3;
      v8f acc[3] = {{}, {}, {}};
      for (int kt = 0; kt < DD / 32; ++kt) {
        v16bf a = load_A_bf(xbf_d, DD, mt * 16, kt * 32, lane);
#pragma unroll
        for (int j = 0; j < 3; ++j) {
          v16bf bfr = load_B_packed(wih_d, G3 / 16, kt, nt0 + j, lane);
          acc[j] = wmma_bf16(a, bfr, acc[j]);
        }
      }
#pragma unroll
      for (int j = 0; j < 3; ++j)
        store_C(gi_d, G3, mt * 16, (nt0 + j) * 16, lane, acc[j]);
    }
    grid_sync(bar, NWG);

    // ---- Phase F: GRU elementwise update + output -----------------------
    for (int i = lwg * 256 + tid; i < BATCH * HID; i += WPD * 256) {
      int b = i / HID, h = i - b * HID;
      float ir = gi_d[(size_t)b * G3 + h] + bih_d[h];
      float iz = gi_d[(size_t)b * G3 + HID + h] + bih_d[HID + h];
      float in_ = gi_d[(size_t)b * G3 + 2 * HID + h] + bih_d[2 * HID + h];
      float hr = gh_d[(size_t)b * G3 + h] + bhh_d[h];
      float hz = gh_d[(size_t)b * G3 + HID + h] + bhh_d[HID + h];
      float hn = gh_d[(size_t)b * G3 + 2 * HID + h] + bhh_d[2 * HID + h];
      float r = fast_sig(ir + hr);
      float z = fast_sig(iz + hz);
      float n = fast_tanh(in_ + r * hn);
      float ns = (1.f - z) * n + z * st_d[i];
      st_d[i] = ns;
      stb_d[i] = (__bf16)ns;
      out[((size_t)p * BATCH + b) * (2 * HID) + dir * HID + h] = ns;
    }
    grid_sync(bar, NWG);
  }
}

// ---------------------------------------------------------------------------
// Host launcher
// ---------------------------------------------------------------------------
extern "C" void kernel_launch(void* const* d_in, const int* in_sizes, int n_in,
                              void* d_out, int out_size, void* d_ws,
                              size_t ws_size, hipStream_t stream) {
  const float* q_input = (const float*)d_in[0];
  const float* p_input = (const float*)d_in[1];
  const unsigned char* q_mask = (const unsigned char*)d_in[2];
  const float* Wq = (const float*)d_in[3];
  const float* Wp = (const float*)d_in[4];
  const float* Ws = (const float*)d_in[5];
  const float* v_w = (const float*)d_in[6];
  const float* gate_W = (const float*)d_in[7];
  const float* gate_b = (const float*)d_in[8];
  const float* gru_Wih = (const float*)d_in[9];
  const float* gru_Whh = (const float*)d_in[10];
  const float* gru_bih = (const float*)d_in[11];
  const float* gru_bhh = (const float*)d_in[12];
  float* out = (float*)d_out;

  char* base = (char*)d_ws;
  size_t off = 0;
  auto take = [&](size_t bytes) -> char* {
    off = (off + 255) & ~(size_t)255;
    char* p = base + off;
    off += bytes;
    return p;
  };
  unsigned* bar = (unsigned*)take(256);
  __bf16* gateWT = (__bf16*)take((size_t)2 * DD * DD * 2);
  __bf16* wihT = (__bf16*)take((size_t)2 * DD * G3 * 2);
  __bf16* whhT = (__bf16*)take((size_t)2 * HID * G3 * 2);
  __bf16* wsB = (__bf16*)take((size_t)2 * HID * HID * 2);
  __bf16* wpB = (__bf16*)take((size_t)2 * PIN * HID * 2);
  __bf16* wqB = (__bf16*)take((size_t)2 * QIN * HID * 2);
  float* qtempB = (float*)take((size_t)2 * QLEN * BATCH * HID * 4);
  float* stempB = (float*)take((size_t)2 * BATCH * HID * 4);
  float* ptmpB = (float*)take((size_t)2 * BATCH * HID * 4);
  float* xpreB = (float*)take((size_t)2 * BATCH * DD * 4);
  __bf16* xprebfB = (__bf16*)take((size_t)2 * BATCH * DD * 2);
  __bf16* xbfB = (__bf16*)take((size_t)2 * BATCH * DD * 2);
  float* ghB = (float*)take((size_t)2 * BATCH * G3 * 4);
  float* giB = (float*)take((size_t)2 * BATCH * G3 * 4);
  float* stateB = (float*)take((size_t)2 * BATCH * HID * 4);
  __bf16* statebfB = (__bf16*)take((size_t)2 * BATCH * HID * 2);

  // deterministic barrier state each call (workspace poisoned by harness)
  hipMemsetAsync(bar, 0, 256, stream);

  auto packB = [&](const float* src, __bf16* dst, int K, int N, int tr) {
    int total = K * N;
    int blocks = (total + 255) / 256;
    for (int dir = 0; dir < 2; ++dir)
      hipLaunchKernelGGL(pack_B_kernel, dim3(blocks), dim3(256), 0, stream,
                         src + (size_t)dir * K * N, dst + (size_t)dir * K * N,
                         K, N, tr);
  };
  packB(gate_W, gateWT, DD, DD, 1);    // gate_W is (j,k): transpose to (k,j)
  packB(gru_Wih, wihT, DD, G3, 1);     // (768,1024) -> (1024,768)
  packB(gru_Whh, whhT, HID, G3, 1);    // (768,256)  -> (256,768)
  packB(Ws, wsB, HID, HID, 0);
  packB(Wp, wpB, PIN, HID, 0);
  packB(Wq, wqB, QIN, HID, 0);

  hipLaunchKernelGGL(qtemp_kernel, dim3(1920), dim3(256), 0, stream, q_input,
                     wqB, qtempB);

  hipLaunchKernelGGL(scan_kernel, dim3(32), dim3(256), 0, stream, q_input,
                     p_input, q_mask, v_w, gate_b, gru_bih, gru_bhh, gateWT,
                     wihT, whhT, wsB, wpB, qtempB, stempB, ptmpB, xpreB,
                     xprebfB, xbfB, ghB, giB, stateB, statebfB, out, bar);

  (void)in_sizes; (void)n_in; (void)out_size; (void)ws_size;
}
